// AttentionMechanism_20040317403413
// MI455X (gfx1250) — matrix operand order repaired and checked by reading, not compiled, not run on hardware
//
#include <hip/hip_runtime.h>

typedef __attribute__((ext_vector_type(16))) _Float16 v16h;
typedef __attribute__((ext_vector_type(4)))  _Float16 v4h;
typedef __attribute__((ext_vector_type(8)))  float    v8f;

#define N_AGENTS 512
#define TRAJ_LEN 16
#define TAU      2048
#define MSG      32768
#define HID      512
#define DV       64
#define SPLITK   8

// ---- WMMA GEMM config: 128x64 C tile, K-step 64, 128 threads (4 wave32) ----
// Each wave computes a 32(M) x 64(N) strip: 2x4 accumulators, 16 WMMAs per K-tile.
#define BM 128
#define BN 64
#define BK 64
#define LDK (BK + 8)   // f16 elems; 144B row stride: 16B aligned, conflict-free

// C[M,N] = A[M,K]*B[K,N] (+bias when gridDim.z==1).
// gridDim.z>1: split-K; block z covers K-range [z*K/Z,(z+1)*K/Z) and writes a
// raw partial tile to C + z*M*N (reduced later, deterministically).
__global__ __launch_bounds__(128) void gemm_f16_wmma(
    const float* __restrict__ A, const float* __restrict__ B,
    const float* __restrict__ bias, float* __restrict__ C,
    int M, int N, int K)
{
    __shared__ _Float16 As[BM][LDK];   // A tile [m][k]
    __shared__ _Float16 Bt[BN][LDK];   // B tile transposed [n][k]

    const int tid  = threadIdx.x;
    const int wave = tid >> 5;
    const int lane = tid & 31;
    const int m0 = blockIdx.x * BM;
    const int n0 = blockIdx.y * BN;

    const int Kc   = K / gridDim.z;
    const int kBeg = blockIdx.z * Kc;
    const int kEnd = kBeg + Kc;

    // ISA 16-bit A 16x32 layout: lanes 0-15 -> K {0..7,16..23}; lanes 16-31 -> K {8..15,24..31}
    const int mrow0 = wave * 32 + (lane & 15);
    const int klo   = (lane >> 4) * 8;
    // ISA 16-bit B 32x16 layout: lane%16 = N col; lanes 0-15 -> K {0..7,16..23}; lanes 16-31 -> K {8..15,24..31}
    const int nrow  = lane & 15;
    const int kbb   = (lane >> 4) * 8;

    v8f acc[2][4];
    #pragma unroll
    for (int mt = 0; mt < 2; ++mt)
        #pragma unroll
        for (int nt = 0; nt < 4; ++nt)
            #pragma unroll
            for (int j = 0; j < 8; ++j) acc[mt][nt][j] = 0.0f;

    for (int kt = kBeg; kt < kEnd; kt += BK) {
        if (kt + BK < kEnd) {   // hint next K-tile into cache (global_prefetch_b8)
            __builtin_prefetch(A + (size_t)(m0 + tid) * K + (kt + BK), 0, 3);
            __builtin_prefetch(B + (size_t)(kt + BK + (tid & 63)) * N + n0 + ((tid >> 6) << 5), 0, 3);
        }

        // ---- stage A: 128x64 f32 -> f16, packed ds_store_b64 ----
        #pragma unroll 4
        for (int i = 0; i < 16; ++i) {
            int idx = i * 128 + tid;            // 2048 float4
            int r = idx >> 4;                   // 0..127
            int c = (idx & 15) << 2;            // 0..60
            float4 f = *(const float4*)(A + (size_t)(m0 + r) * K + (kt + c));
            v4h h = { (_Float16)f.x, (_Float16)f.y, (_Float16)f.z, (_Float16)f.w };
            *(v4h*)&As[r][c] = h;
        }
        // ---- stage B transposed via 4(k)x4(n) micro-blocks, packed b64 stores ----
        #pragma unroll
        for (int i = 0; i < 2; ++i) {
            int sb = i * 128 + tid;             // 256 micro-blocks
            int kq = (sb >> 4) << 2;            // k base 0..60
            int cq = (sb & 15) << 2;            // n base 0..60
            const float* bp = B + (size_t)(kt + kq) * N + (n0 + cq);
            float4 f0 = *(const float4*)(bp);
            float4 f1 = *(const float4*)(bp + N);
            float4 f2 = *(const float4*)(bp + 2 * (size_t)N);
            float4 f3 = *(const float4*)(bp + 3 * (size_t)N);
            v4h h0 = { (_Float16)f0.x, (_Float16)f1.x, (_Float16)f2.x, (_Float16)f3.x };
            v4h h1 = { (_Float16)f0.y, (_Float16)f1.y, (_Float16)f2.y, (_Float16)f3.y };
            v4h h2 = { (_Float16)f0.z, (_Float16)f1.z, (_Float16)f2.z, (_Float16)f3.z };
            v4h h3 = { (_Float16)f0.w, (_Float16)f1.w, (_Float16)f2.w, (_Float16)f3.w };
            *(v4h*)&Bt[cq + 0][kq] = h0;
            *(v4h*)&Bt[cq + 1][kq] = h1;
            *(v4h*)&Bt[cq + 2][kq] = h2;
            *(v4h*)&Bt[cq + 3][kq] = h3;
        }
        __syncthreads();

        // ---- two K=32 WMMA steps; each B fragment feeds 2 M-subtiles ----
        #pragma unroll
        for (int ks = 0; ks < 2; ++ks) {
            v16h a0, a1;
            const _Float16* ap0 = &As[mrow0][ks * 32 + klo];
            const _Float16* ap1 = &As[mrow0 + 16][ks * 32 + klo];
            #pragma unroll
            for (int j = 0; j < 8; ++j) {
                a0[j] = ap0[j]; a0[j + 8] = ap0[j + 16];
                a1[j] = ap1[j]; a1[j + 8] = ap1[j + 16];
            }
            #pragma unroll
            for (int nt = 0; nt < 4; ++nt) {
                v16h b;
                const _Float16* bp = &Bt[nt * 16 + nrow][ks * 32 + kbb];
                #pragma unroll
                for (int j = 0; j < 8; ++j) { b[j] = bp[j]; b[j + 8] = bp[j + 16]; }
                acc[0][nt] = __builtin_amdgcn_wmma_f32_16x16x32_f16(
                    false, a0, false, b, (short)0, acc[0][nt], false, false);
                acc[1][nt] = __builtin_amdgcn_wmma_f32_16x16x32_f16(
                    false, a1, false, b, (short)0, acc[1][nt], false, false);
            }
        }
        __syncthreads();
    }

    // ---- epilogue: C 16x16 f32 layout: VGPR r -> row (lane>=16 ? 8+r : r), col lane%16 ----
    float* Cout = C + (size_t)blockIdx.z * M * N;
    const bool withBias = (gridDim.z == 1);
    #pragma unroll
    for (int mt = 0; mt < 2; ++mt) {
        #pragma unroll
        for (int nt = 0; nt < 4; ++nt) {
            int col = n0 + nt * 16 + (lane & 15);
            float bcol = withBias ? bias[col] : 0.0f;
            #pragma unroll
            for (int r = 0; r < 8; ++r) {
                int row = m0 + wave * 32 + mt * 16 + (lane >> 4) * 8 + r;
                Cout[(size_t)row * N + col] = acc[mt][nt][r] + bcol;
            }
        }
    }
}

// Deterministic split-K reduction: C[idx] = bias[col] + sum_z part[z][idx]
__global__ __launch_bounds__(256) void splitk_reduce(
    const float* __restrict__ part, const float* __restrict__ bias,
    float* __restrict__ C, int M, int N, int S)
{
    int idx = blockIdx.x * 256 + threadIdx.x;
    if (idx >= M * N) return;
    float s = bias[idx % N];
    for (int z = 0; z < S; ++z) s += part[(size_t)z * M * N + idx];
    C[idx] = s;
}

// Per-agent attention: one wave32 per agent; 8 waves (256 thr) per block.
__global__ __launch_bounds__(256) void attn_softmax_av(
    const float* __restrict__ q,   // [N_AGENTS, HID]
    const float* __restrict__ k,   // [TRAJ_LEN, N_AGENTS, HID]
    const float* __restrict__ v,   // [TRAJ_LEN, N_AGENTS, DV]
    float* __restrict__ out)       // [N_AGENTS, DV]
{
    const int lane = threadIdx.x & 31;
    const int wave = threadIdx.x >> 5;
    const int n = blockIdx.x * 8 + wave;
    const float scale = 0.04419417382415922f;   // 1/sqrt(512)

    const float* qp = q + (size_t)n * HID;
    float s[TRAJ_LEN];
    #pragma unroll
    for (int t = 0; t < TRAJ_LEN; ++t) {
        const float* kp = k + ((size_t)t * N_AGENTS + n) * HID;
        float p = 0.0f;
        #pragma unroll 4
        for (int h = lane; h < HID; h += 32) p = fmaf(qp[h], kp[h], p);
        #pragma unroll
        for (int off = 16; off > 0; off >>= 1) p += __shfl_xor(p, off, 32);
        s[t] = p * scale;                       // all lanes hold the full sum
    }
    float mx = s[0];
    #pragma unroll
    for (int t = 1; t < TRAJ_LEN; ++t) mx = fmaxf(mx, s[t]);
    float denom = 0.0f;
    #pragma unroll
    for (int t = 0; t < TRAJ_LEN; ++t) { s[t] = __expf(s[t] - mx); denom += s[t]; }
    const float inv = 1.0f / denom;

    #pragma unroll
    for (int d0 = 0; d0 < DV; d0 += 32) {
        int d = d0 + lane;
        float o = 0.0f;
        #pragma unroll
        for (int t = 0; t < TRAJ_LEN; ++t)
            o = fmaf(s[t], v[((size_t)t * N_AGENTS + n) * DV + d], o);
        out[(size_t)n * DV + d] = o * inv;
    }
}

extern "C" void kernel_launch(void* const* d_in, const int* in_sizes, int n_in,
                              void* d_out, int out_size, void* d_ws, size_t ws_size,
                              hipStream_t stream) {
    (void)in_sizes; (void)n_in; (void)out_size; (void)ws_size;
    const float* traj = (const float*)d_in[0];  // [16, 512, 2048] == [8192, 2048]
    const float* rmsg = (const float*)d_in[1];  // [512, 32768]
    const float* Wq   = (const float*)d_in[2];  // [32768, 512]
    const float* bq   = (const float*)d_in[3];  // [512]
    const float* Wk   = (const float*)d_in[4];  // [2048, 512]
    const float* bk   = (const float*)d_in[5];  // [512]
    const float* Wv   = (const float*)d_in[6];  // [2048, 64]
    const float* bv   = (const float*)d_in[7];  // [64]
    float* out = (float*)d_out;

    // workspace: q (1MB) | k (16MB) | v (2MB); split-K partials (8MB) alias the
    // k/v region — they are fully consumed by splitk_reduce before GEMM2 runs.
    float* qbuf  = (float*)d_ws;                              // 512*512
    float* kbuf  = qbuf + (size_t)N_AGENTS * HID;             // 8192*512
    float* vbuf  = kbuf + (size_t)TRAJ_LEN * N_AGENTS * HID;  // 8192*64
    float* qpart = kbuf;                                      // SPLITK * 512*512

    const int TN = TRAJ_LEN * N_AGENTS; // 8192

    // q = rmsg @ Wq + bq  [512 x 512], K=32768, split-K=8 -> 256 blocks
    gemm_f16_wmma<<<dim3(N_AGENTS / BM, HID / BN, SPLITK), 128, 0, stream>>>(
        rmsg, Wq, bq, qpart, N_AGENTS, HID, MSG);
    splitk_reduce<<<(N_AGENTS * HID) / 256, 256, 0, stream>>>(
        qpart, bq, qbuf, N_AGENTS, HID, SPLITK);
    // k = taus @ Wk + bk  [8192 x 512], K=2048 -> 512 blocks
    gemm_f16_wmma<<<dim3(TN / BM, HID / BN, 1), 128, 0, stream>>>(
        traj, Wk, bk, kbuf, TN, HID, TAU);
    // v = taus @ Wv + bv  [8192 x 64], K=2048 -> 64 blocks
    gemm_f16_wmma<<<dim3(TN / BM, DV / BN, 1), 128, 0, stream>>>(
        traj, Wv, bv, vbuf, TN, DV, TAU);
    // per-agent softmax attention
    attn_softmax_av<<<N_AGENTS / 8, 256, 0, stream>>>(qbuf, kbuf, vbuf, out);
}
